// Attention_24696061952715
// MI455X (gfx1250) — matrix operand order repaired and checked by reading, not compiled, not run on hardware
//
#include <hip/hip_runtime.h>
#include <hip/hip_bf16.h>

// ---------------------------------------------------------------------------
// Cross-attention for MI455X (gfx1250, wave32, WMMA).
//   B=4, SQ=SKV=1024, E=1024, H=16, D=64
// Pipeline:
//   1) proj_bf16_kernel<TR> x3 : Q/K/V = X@W + b, f32 in -> bf16 out.
//        Q,K stored head-split [B,H,S,D]; V stored TRANSPOSED [B,H,D,S] so the
//        attention PV B-operand is a contiguous 32-byte global load.
//   2) attn_kernel             : fused flash attention; K tile staged per-block
//        via GLOBAL_LOAD_ASYNC_TO_LDS_B128 (ASYNCcnt); scores/PV/row-sums via
//        v_wmma_f32_16x16x32_bf16; row-max via ds_swizzle_b32 xor reductions.
// Workspace: 3 * B*H*S*D bf16 planes = 24 MB
// ---------------------------------------------------------------------------

typedef __attribute__((ext_vector_type(16))) __bf16 v16bf;
typedef __attribute__((ext_vector_type(8)))  __bf16 v8bf;
typedef __attribute__((ext_vector_type(8)))  float  v8f;

#define WMMA_BF16(a, b, c) \
  __builtin_amdgcn_wmma_f32_16x16x32_bf16(false, (a), false, (b), (short)0, (c), false, false)

constexpr int Bc   = 4;
constexpr int SQc  = 1024;
constexpr int SKVc = 1024;
constexpr int Ec   = 1024;
constexpr int Hc   = 16;
constexpr int Dc   = 64;

// Async copy of 16 bytes global -> LDS (per lane), tracked by ASYNCcnt.
__device__ __forceinline__ void async_copy_b128(unsigned lds_byte_off, const void* gaddr) {
  asm volatile("global_load_async_to_lds_b128 %0, %1, off"
               :: "v"(lds_byte_off), "v"(gaddr)
               : "memory");
}
__device__ __forceinline__ void wait_async0() {
  asm volatile("s_wait_asynccnt 0x0" ::: "memory");
}

// xor-swizzle within 16-lane groups: single ds_swizzle_b32, no index VALU.
// group-of-32 mode: and_mask=0x1f, or_mask=0, xor_mask=MASK.
template <int MASK>
__device__ __forceinline__ float swz_xor(float x) {
  int i = __builtin_amdgcn_ds_swizzle(__builtin_bit_cast(int, x),
                                      (MASK << 10) | 0x1f);
  return __builtin_bit_cast(float, i);
}

// ---------------------------------------------------------------------------
// Projection GEMM: block = 8 waves -> 128 rows x 128 cols tile.
// Each wave: 32 rows x 64 cols = 8 WMMAs per K-step sharing one B fragment.
// TR=0 : Y[((b*H+h)*S+s)*D+d]   (Q, K)
// TR=1 : Y[((b*H+h)*D+d)*S+s]   (V, pre-transposed for attention)
// ---------------------------------------------------------------------------
template <int TR>
__global__ __launch_bounds__(256)
void proj_bf16_kernel(const float* __restrict__ X, const float* __restrict__ W,
                      const float* __restrict__ bias, __bf16* __restrict__ Y) {
  __shared__ __bf16 WtLds[128 * 32];   // [n][k], 8 KB

  const int tid  = threadIdx.x;
  const int lm   = tid & 15;
  const int half = (tid >> 4) & 1;
  const int wid  = tid >> 5;
  const int wr   = wid & 3;            // row strip (32 rows each)
  const int wc   = wid >> 2;           // col group (64 cols each)

  const int rowBase = blockIdx.x * 128 + wr * 32;
  const int colBase = blockIdx.y * 128;

  v8f acc[2][4] = {};
  const float* xrow0 = X + (size_t)(rowBase + lm) * Ec;
  const float* xrow1 = X + (size_t)(rowBase + 16 + lm) * Ec;

  for (int kb = 0; kb < Ec; kb += 32) {
    // cooperative load of W[kb..kb+32][colBase..+128], transposed, f32 -> bf16
#pragma unroll
    for (int i = 0; i < 16; ++i) {
      int idx = tid + i * 256;
      int n = idx & 127, k = idx >> 7;
      WtLds[n * 32 + k] = (__bf16)W[(size_t)(kb + k) * Ec + colBase + n];
    }
    __syncthreads();

    // two A fragments (rows lm and lm+16), contiguous k order (k = 16*half + e)
    v16bf a0, a1;
#pragma unroll
    for (int e = 0; e < 8; ++e) {
      a0[e]     = (__bf16)xrow0[kb + half * 16 + e];
      a0[8 + e] = (__bf16)xrow0[kb + half * 16 + 8 + e];
      a1[e]     = (__bf16)xrow1[kb + half * 16 + e];
      a1[8 + e] = (__bf16)xrow1[kb + half * 16 + 8 + e];
    }

#pragma unroll
    for (int nc = 0; nc < 4; ++nc) {
      v16bf b = *(const v16bf*)&WtLds[(wc * 64 + nc * 16 + lm) * 32 + half * 16];
      acc[0][nc] = WMMA_BF16(a0, b, acc[0][nc]);
      acc[1][nc] = WMMA_BF16(a1, b, acc[1][nc]);
    }
    __syncthreads();
  }

  // epilogue: bias add, bf16 convert, head-split (optionally transposed) store
#pragma unroll
  for (int rs = 0; rs < 2; ++rs) {
#pragma unroll
    for (int nc = 0; nc < 4; ++nc) {
      int n  = colBase + wc * 64 + nc * 16 + lm;
      float bi = bias[n];
      int h = n >> 6, d = n & 63;
#pragma unroll
      for (int r = 0; r < 8; ++r) {
        int row = rowBase + rs * 16 + r + 8 * half;
        int b_  = row >> 10;
        int s   = row & 1023;
        size_t o;
        if (TR) o = (((size_t)b_ * Hc + h) * Dc + d) * SQc + s;
        else    o = (((size_t)b_ * Hc + h) * SQc + s) * Dc + d;
        Y[o] = (__bf16)(acc[rs][nc][r] + bi);
      }
    }
  }
}

// ---------------------------------------------------------------------------
// Fused flash attention. Block = (b,h) x 128 q rows; wave = 16 q rows.
// Per 32-wide KV tile:
//   - K tile (32x64 bf16 = 4KB) staged once per block via async-to-LDS B128
//   - scores: 4 WMMAs from KLds
//   - row max: ds_swizzle xor reduction; row SUM: extra WMMA against an
//     all-ones B fragment (denominator lives in a C fragment, lane-replicated)
//   - probs -> per-wave LDS patch -> A layout
//   - PV: 4 WMMAs, V B-operand read directly from pre-transposed global plane
// ---------------------------------------------------------------------------
__global__ __launch_bounds__(256)
void attn_kernel(const __bf16* __restrict__ Qp, const __bf16* __restrict__ Kp,
                 const __bf16* __restrict__ Vtp, const float* __restrict__ mask,
                 float* __restrict__ Out) {
  __shared__ __bf16 KLds[32 * 64];       // K tile [kv][d], 4 KB
  __shared__ __bf16 pLds[8 * 16 * 32];   // per-wave prob staging, 8 KB

  const int tid  = threadIdx.x;
  const int lm   = tid & 15;
  const int half = (tid >> 4) & 1;
  const int wid  = tid >> 5;

  const int bh = blockIdx.y;             // b*H + h
  const int b_ = bh >> 4, h = bh & 15;
  const int qrow0 = blockIdx.x * 128 + wid * 16;

  const __bf16* qplane  = Qp  + (size_t)bh * SQc * Dc;
  const __bf16* kplane  = Kp  + (size_t)bh * SKVc * Dc;
  const __bf16* vtplane = Vtp + (size_t)bh * Dc * SKVc;   // [d][s]
  const float*  mrow    = mask + (size_t)b_ * SKVc;

  // Q fragments for this wave's 16 rows, D=64 -> two K=32 A fragments
  v16bf qf[2];
  const __bf16* qr = qplane + (size_t)(qrow0 + lm) * Dc;
#pragma unroll
  for (int ds = 0; ds < 2; ++ds) {
    v8bf lo = *(const v8bf*)(qr + ds * 32 + half * 16);
    v8bf hi = *(const v8bf*)(qr + ds * 32 + half * 16 + 8);
#pragma unroll
    for (int e = 0; e < 8; ++e) { qf[ds][e] = lo[e]; qf[ds][8 + e] = hi[e]; }
  }

  // constant all-ones B fragment: rowsum(P) = P @ ones
  v16bf onesb;
#pragma unroll
  for (int e = 0; e < 16; ++e) onesb[e] = (__bf16)1.0f;

  v8f acc[4] = {};
  v8f lacc   = {};                       // softmax denominator, lane-replicated
  float mrun[8];
#pragma unroll
  for (int r = 0; r < 8; ++r) mrun[r] = -1e30f;

  __bf16* pw = &pLds[wid * 16 * 32];

  // per-thread source/dest for the async K-tile copy (16B per thread)
  const int kvl = tid >> 3;              // 0..31
  const int dch = tid & 7;               // 8-element chunk along D
  const unsigned kldsOff =
      (unsigned)(uintptr_t)&KLds[kvl * 64 + dch * 8];

  for (int kv = 0; kv < SKVc; kv += 32) {
    // stage K tile: one async B128 copy per thread, drained with ASYNCcnt
    async_copy_b128(kldsOff, kplane + (size_t)(kv + kvl) * Dc + dch * 8);
    if (kv + 32 < SKVc) {  // hint next tiles (global_prefetch_b8)
      __builtin_prefetch(kplane + (size_t)(kv + 32 + kvl) * Dc, 0, 1);
      __builtin_prefetch(vtplane + (size_t)(lm) * SKVc + kv + 32, 0, 1);
    }
    wait_async0();
    __syncthreads();

    // scores: two 16-wide kv column groups, B operand from KLds
    v8f sf[2];
#pragma unroll
    for (int g = 0; g < 2; ++g) {
      const __bf16* krow = &KLds[(g * 16 + lm) * 64];
      v16bf bk0 = *(const v16bf*)(krow + half * 16);
      v16bf bk1 = *(const v16bf*)(krow + 32 + half * 16);
      v8f s = {};
      s = WMMA_BF16(qf[0], bk0, s);
      s = WMMA_BF16(qf[1], bk1, s);
      float mk = mrow[kv + g * 16 + lm];
#pragma unroll
      for (int r = 0; r < 8; ++r) s[r] = s[r] * 0.125f + mk;  // 1/sqrt(64)
      sf[g] = s;
    }

    // online softmax: row max via ds_swizzle xor-reduce in the 16-lane half
    float scale[8];
#pragma unroll
    for (int r = 0; r < 8; ++r) {
      float v = fmaxf(sf[0][r], sf[1][r]);
      v = fmaxf(v, swz_xor<1>(v));
      v = fmaxf(v, swz_xor<2>(v));
      v = fmaxf(v, swz_xor<4>(v));
      v = fmaxf(v, swz_xor<8>(v));
      float mnew = fmaxf(mrun[r], v);
      scale[r] = __expf(mrun[r] - mnew);
      sf[0][r] = __expf(sf[0][r] - mnew);
      sf[1][r] = __expf(sf[1][r] - mnew);
      mrun[r] = mnew;
    }
#pragma unroll
    for (int nc = 0; nc < 4; ++nc)
#pragma unroll
      for (int r = 0; r < 8; ++r) acc[nc][r] *= scale[r];
#pragma unroll
    for (int r = 0; r < 8; ++r) lacc[r] *= scale[r];

    // probs: C layout -> LDS -> A layout (wave-local, DS ops are in-order)
#pragma unroll
    for (int g = 0; g < 2; ++g)
#pragma unroll
      for (int r = 0; r < 8; ++r)
        pw[(r + 8 * half) * 32 + g * 16 + lm] = (__bf16)sf[g][r];

    v16bf ap;
    {
      const __bf16* pr = pw + lm * 32;
      v8bf lo = *(const v8bf*)(pr + half * 16);
      v8bf hi = *(const v8bf*)(pr + half * 16 + 8);
#pragma unroll
      for (int e = 0; e < 8; ++e) { ap[e] = lo[e]; ap[8 + e] = hi[e]; }
    }

    // denominator: rowsum(P) accumulated on the matrix pipe
    lacc = WMMA_BF16(ap, onesb, lacc);

    // PV: V B-operand straight from pre-transposed global plane (32B loads)
#pragma unroll
    for (int nc = 0; nc < 4; ++nc) {
      v16bf bv = *(const v16bf*)(vtplane + (size_t)(nc * 16 + lm) * SKVc
                                 + kv + half * 16);
      acc[nc] = WMMA_BF16(ap, bv, acc[nc]);
    }
    __syncthreads();   // protect KLds before next tile's async overwrite
  }

  // normalize and store: output is [B, SQ, H, D] flattened
#pragma unroll
  for (int nc = 0; nc < 4; ++nc) {
    int d = nc * 16 + lm;
#pragma unroll
    for (int r = 0; r < 8; ++r) {
      int q = qrow0 + r + 8 * half;
      size_t o = (((size_t)b_ * SQc + q) * Hc + h) * Dc + d;
      Out[o] = acc[nc][r] / lacc[r];
    }
  }
}

// ---------------------------------------------------------------------------
extern "C" void kernel_launch(void* const* d_in, const int* in_sizes, int n_in,
                              void* d_out, int out_size, void* d_ws, size_t ws_size,
                              hipStream_t stream) {
  (void)in_sizes; (void)n_in; (void)out_size; (void)ws_size;
  const float* hidden  = (const float*)d_in[0];
  const float* context = (const float*)d_in[1];
  const float* mask    = (const float*)d_in[2];
  const float* Wq      = (const float*)d_in[3];
  const float* bq      = (const float*)d_in[4];
  const float* Wk      = (const float*)d_in[5];
  const float* bk      = (const float*)d_in[6];
  const float* Wv      = (const float*)d_in[7];
  const float* bv      = (const float*)d_in[8];
  float* out = (float*)d_out;

  const size_t plane = (size_t)Bc * Hc * SQc * Dc;   // elements per bf16 plane
  __bf16* qws = (__bf16*)d_ws;
  __bf16* kws = qws + plane;
  __bf16* vws = kws + plane;                          // stored [B,H,D,S]

  dim3 pg(Bc * SQc / 128, Ec / 128);                  // (32, 8)
  proj_bf16_kernel<0><<<pg, 256, 0, stream>>>(hidden,  Wq, bq, qws);
  proj_bf16_kernel<0><<<pg, 256, 0, stream>>>(context, Wk, bk, kws);
  proj_bf16_kernel<1><<<pg, 256, 0, stream>>>(context, Wv, bv, vws);

  dim3 ag(SQc / 128, Bc * Hc);                        // (8, 64)
  attn_kernel<<<ag, 256, 0, stream>>>(qws, kws, vws, mask, out);
}
